// ODE_RNN_58832462021259
// MI455X (gfx1250) — matrix operand order repaired and checked by reading, not compiled, not run on hardware
//
#include <hip/hip_runtime.h>
#include <hip/hip_bf16.h>
#include <math.h>

// ---------------------------------------------------------------------------
// ODE-RNN persistent-kernel for MI455X (gfx1250, wave32, WMMA).
//
// Sequential scan -> one persistent kernel, 16 WGs x 256 threads (128 waves).
// Weights converted ONCE per launch to bf16 WMMA A-fragment tile order
// (~50 MB, resident in the 192 MB L2 -> HBM touched once).  Each 2048-wide
// matvec phase: stage vector to LDS as bf16 (broadcast ds reads), each wave
// computes a 16-row tile with 64 x v_wmma_f32_16x16x32_bf16, dual
// accumulators + depth-2 load rotation, fused bias+tanh writeout.
// A per-tile compiler memory barrier stops the backend from hoisting all 64
// LDS B-fragments across tiles (which caused >256-VGPR scratch spills).
// Grid barrier = monotonic atomic counter (memset per launch, capture-safe).
// ---------------------------------------------------------------------------

#define HDIM   2048
#define GDIM   6144          // 3*H gate rows
#define INDIM  64
#define OUTDIM 64
#define NSTEP  500
#define KB     (HDIM / 32)   // 64 WMMA K-blocks per 2048-wide matvec
#define NWG    16
#define TPB    256
#define NTHR   (NWG * TPB)   // 4096
#define NWAVES (NTHR / 32)   // 128

typedef __bf16 v16bf __attribute__((ext_vector_type(16)));
typedef float  v8f   __attribute__((ext_vector_type(8)));

// ---- workspace layout (bytes) ---------------------------------------------
constexpr size_t OFF_CNT = 0;                                   // barrier ctr
constexpr size_t OFF_W1  = 256;
constexpr size_t SZ_HH   = (size_t)HDIM * HDIM * 2;             // bf16 bytes
constexpr size_t OFF_W2  = OFF_W1 + SZ_HH;
constexpr size_t OFF_WHH = OFF_W2 + SZ_HH;
constexpr size_t OFF_WHG = OFF_WHH + SZ_HH;
constexpr size_t OFF_WD  = OFF_WHG + (size_t)GDIM * HDIM * 2;
constexpr size_t OFF_VEC = OFF_WD + (size_t)OUTDIM * HDIM * 2;  // f32 area

__device__ __forceinline__ unsigned short f2bf(float f) {
  unsigned u = __float_as_uint(f);
  unsigned r = u + 0x7FFFu + ((u >> 16) & 1u);   // round-to-nearest-even
  return (unsigned short)(r >> 16);
}

// Convert row-major f32 weights -> bf16 tiles in WMMA A-fragment order.
// Block = one 16x32 A fragment (512 bf16, 1KB), lane-major so a wave loads
// its fragment as one fully coalesced 32B-per-lane chunk.
__device__ void conv_tiled(const float* __restrict__ W,
                           unsigned short* __restrict__ Wt,
                           int ntiles, int gtid) {
  const size_t total = (size_t)ntiles * KB * 512;
  for (size_t d = gtid; d < total; d += NTHR) {
    size_t blk = d >> 9;            // /512
    int r    = (int)(d & 511);
    int tile = (int)(blk >> 6);     // /KB
    int kb   = (int)(blk & (KB - 1));
    int lane = r >> 4;
    int e    = r & 15;
    int row  = tile * 16 + (lane & 15);
    // contiguous K order: half=lane>>4 selects K+16; e indexes K within half
    int col  = kb * 32 + ((lane >> 4) << 4) + e;
    Wt[d] = f2bf(W[(size_t)row * HDIM + col]);
  }
}

// Device-wide barrier: monotonic counter, agent-scope acq/rel.
__device__ __forceinline__ void gbar(unsigned* cnt, unsigned* s_epoch) {
  __syncthreads();
  if (threadIdx.x == 0) {
    unsigned e = ++(*s_epoch);
    __hip_atomic_fetch_add(cnt, 1u, __ATOMIC_RELEASE, __HIP_MEMORY_SCOPE_AGENT);
    while (__hip_atomic_load(cnt, __ATOMIC_ACQUIRE, __HIP_MEMORY_SCOPE_AGENT) <
           e * (unsigned)NWG)
      __builtin_amdgcn_s_sleep(2);
  }
  __syncthreads();
}

// Stage y = a (+ c*b) into LDS as bf16 (B operand source for the matvec).
__device__ __forceinline__ void stage(unsigned short* sh,
                                      const float* __restrict__ a,
                                      const float* __restrict__ b, float c) {
  for (int i = threadIdx.x; i < HDIM; i += TPB) {
    float v = a[i];
    if (b) v += c * b[i];
    sh[i] = f2bf(v);
  }
  __syncthreads();
}

// y[tile rows] = act(Wt @ sh + bias).  One 16-row tile per wave iteration,
// K=2048 as 64 x v_wmma_f32_16x16x32_bf16.  B is the staged vector broadcast
// into all 16 N columns (lanes 0-15 read K 0..15, lanes 16-31 read K 16..31
// of each block from LDS -> broadcast, conflict-free), so every column of
// the f32 accumulator holds the matvec result; lanes 0/16 write rows 0-7 /
// 8-15.
__device__ __forceinline__ void matvec2048(const unsigned short* __restrict__ Wt,
                                           int ntiles,
                                           const float* __restrict__ bias,
                                           float* __restrict__ y,
                                           const unsigned short* sh,
                                           int gwave, int lane, int act) {
  const int hsel = (lane < 16) ? 0 : 16;
  for (int tile = gwave; tile < ntiles; tile += NWAVES) {
    // Stop the backend from hoisting the (tile-invariant) LDS B-fragments
    // out of this loop: with KB=64 frags that exceeds 256 VGPRs and spills
    // to scratch.  Re-reading broadcast LDS per tile is essentially free.
    asm volatile("" ::: "memory");

    const unsigned short* base =
        Wt + (size_t)tile * (KB * 512) + (size_t)lane * 16;
    if (tile + NWAVES < ntiles)      // uniform: prefetch this wave's next tile
      __builtin_prefetch((const void*)(base + (size_t)NWAVES * (KB * 512)), 0, 1);

    v8f acc0 = {0.f, 0.f, 0.f, 0.f, 0.f, 0.f, 0.f, 0.f};
    v8f acc1 = {0.f, 0.f, 0.f, 0.f, 0.f, 0.f, 0.f, 0.f};
    v16bf a0 = *(const v16bf*)(base);
    v16bf b0 = *(const v16bf*)(sh + hsel);
#pragma unroll 4
    for (int kb = 0; kb < KB - 1; ++kb) {
      v16bf a1 = *(const v16bf*)(base + (size_t)(kb + 1) * 512);
      v16bf b1 = *(const v16bf*)(sh + (kb + 1) * 32 + hsel);
      if (kb & 1)
        acc1 = __builtin_amdgcn_wmma_f32_16x16x32_bf16(
            false, a0, false, b0, (short)0, acc1, false, false);
      else
        acc0 = __builtin_amdgcn_wmma_f32_16x16x32_bf16(
            false, a0, false, b0, (short)0, acc0, false, false);
      a0 = a1;
      b0 = b1;
    }
    acc1 = __builtin_amdgcn_wmma_f32_16x16x32_bf16(
        false, a0, false, b0, (short)0, acc1, false, false);
    acc0 = acc0 + acc1;

    if (lane == 0 || lane == 16) {
      int rbase = tile * 16 + ((lane == 16) ? 8 : 0);
#pragma unroll
      for (int r = 0; r < 8; ++r) {
        float v = acc0[r] + bias[rbase + r];
        if (act) v = tanhf(v);
        y[rbase + r] = v;
      }
    }
  }
}

__device__ __forceinline__ float sigm(float x) { return 1.f / (1.f + expf(-x)); }

__global__ void __launch_bounds__(TPB, 2)   // cap at 256 VGPRs/wave (no MSB)
ode_rnn_kernel(const float* __restrict__ tarr, const float* __restrict__ x,
               const float* __restrict__ W1, const float* __restrict__ b1,
               const float* __restrict__ W2, const float* __restrict__ b2,
               const float* __restrict__ Wh, const float* __restrict__ bh,
               const float* __restrict__ Wd, const float* __restrict__ bd,
               const float* __restrict__ Wxg, const float* __restrict__ Whg,
               const float* __restrict__ bxg, const float* __restrict__ bhg,
               float* __restrict__ out, unsigned char* __restrict__ ws) {
  unsigned*       cnt  = (unsigned*)(ws + OFF_CNT);
  unsigned short* W1t  = (unsigned short*)(ws + OFF_W1);
  unsigned short* W2t  = (unsigned short*)(ws + OFF_W2);
  unsigned short* Wht  = (unsigned short*)(ws + OFF_WHH);
  unsigned short* Whgt = (unsigned short*)(ws + OFF_WHG);
  unsigned short* Wdt  = (unsigned short*)(ws + OFF_WD);
  float* h   = (float*)(ws + OFF_VEC);
  float* u   = h + HDIM;
  float* k1  = u + HDIM;
  float* k2  = k1 + HDIM;
  float* k3  = k2 + HDIM;
  float* k4  = k3 + HDIM;
  float* tmp = k4 + HDIM;
  float* gh  = tmp + HDIM;
  float* gxa = gh + GDIM;                         // [NSTEP, GDIM]

  const int gtid  = blockIdx.x * TPB + threadIdx.x;
  const int lane  = threadIdx.x & 31;             // wave32
  const int gwave = gtid >> 5;

  __shared__ unsigned s_epoch;
  __shared__ unsigned short sh[HDIM] __attribute__((aligned(32)));
  if (threadIdx.x == 0) s_epoch = 0;

  // ---- per-launch setup: bf16 tiled weights + hoisted Wxg@x_i + h0 --------
  conv_tiled(W1, W1t, HDIM / 16, gtid);
  conv_tiled(W2, W2t, HDIM / 16, gtid);
  conv_tiled(Wh, Wht, HDIM / 16, gtid);
  conv_tiled(Whg, Whgt, GDIM / 16, gtid);
  conv_tiled(Wd, Wdt, OUTDIM / 16, gtid);
  for (size_t idx = gtid; idx < (size_t)NSTEP * GDIM; idx += NTHR) {
    int n = (int)(idx / GDIM), row = (int)(idx % GDIM);
    float acc = bxg[row];
    const float* wr = Wxg + (size_t)row * INDIM;
    const float* xv = x + (size_t)n * INDIM;
#pragma unroll 8
    for (int k = 0; k < INDIM; ++k) acc = fmaf(wr[k], xv[k], acc);
    gxa[idx] = acc;
  }
  for (int i = gtid; i < HDIM; i += NTHR) h[i] = 0.f;
  gbar(cnt, &s_epoch);

  // ---- sequential scan ----------------------------------------------------
  float tprev = 0.f;
  for (int n = 0; n < NSTEP; ++n) {
    float tn = tarr[n];
    float dt = (tn - tprev) * 0.25f;              // N_SUB=4 substeps
    tprev = tn;
    const float dt2 = 0.5f * dt, dt6 = dt * (1.f / 6.f);

    for (int s = 0; s < 4; ++s) {
      // k1 = f(h)
      stage(sh, h, nullptr, 0.f);
      matvec2048(W1t, HDIM / 16, b1, u, sh, gwave, lane, 1);
      gbar(cnt, &s_epoch);
      stage(sh, u, nullptr, 0.f);
      matvec2048(W2t, HDIM / 16, b2, k1, sh, gwave, lane, 0);
      gbar(cnt, &s_epoch);
      // k2 = f(h + dt/2*k1)
      stage(sh, h, k1, dt2);
      matvec2048(W1t, HDIM / 16, b1, u, sh, gwave, lane, 1);
      gbar(cnt, &s_epoch);
      stage(sh, u, nullptr, 0.f);
      matvec2048(W2t, HDIM / 16, b2, k2, sh, gwave, lane, 0);
      gbar(cnt, &s_epoch);
      // k3 = f(h + dt/2*k2)
      stage(sh, h, k2, dt2);
      matvec2048(W1t, HDIM / 16, b1, u, sh, gwave, lane, 1);
      gbar(cnt, &s_epoch);
      stage(sh, u, nullptr, 0.f);
      matvec2048(W2t, HDIM / 16, b2, k3, sh, gwave, lane, 0);
      gbar(cnt, &s_epoch);
      // k4 = f(h + dt*k3)
      stage(sh, h, k3, dt);
      matvec2048(W1t, HDIM / 16, b1, u, sh, gwave, lane, 1);
      gbar(cnt, &s_epoch);
      stage(sh, u, nullptr, 0.f);
      matvec2048(W2t, HDIM / 16, b2, k4, sh, gwave, lane, 0);
      gbar(cnt, &s_epoch);
      // h += dt/6 * (k1 + 2k2 + 2k3 + k4)
      for (int i = gtid; i < HDIM; i += NTHR)
        h[i] += dt6 * (k1[i] + 2.f * k2[i] + 2.f * k3[i] + k4[i]);
      gbar(cnt, &s_epoch);
    }

    // head: tmp = Wh@h+bh ; gates: gh = Whg@h+bhg  (both read staged h)
    stage(sh, h, nullptr, 0.f);
    matvec2048(Wht, HDIM / 16, bh, tmp, sh, gwave, lane, 0);
    matvec2048(Whgt, GDIM / 16, bhg, gh, sh, gwave, lane, 0);
    gbar(cnt, &s_epoch);

    // out_n = tanh(Wd@tmp + bd)  -> d_out[n*64 ..]
    stage(sh, tmp, nullptr, 0.f);
    matvec2048(Wdt, OUTDIM / 16, bd, out + (size_t)n * OUTDIM, sh, gwave, lane, 1);

    // GRU (torch gate order r,z,n), gx hoisted: h = (1-z)*nn + z*h
    const float* gx = gxa + (size_t)n * GDIM;
    for (int i = gtid; i < HDIM; i += NTHR) {
      float r  = sigm(gx[i] + gh[i]);
      float z  = sigm(gx[HDIM + i] + gh[HDIM + i]);
      float nn = tanhf(gx[2 * HDIM + i] + r * gh[2 * HDIM + i]);
      h[i] = (1.f - z) * nn + z * h[i];
    }
    gbar(cnt, &s_epoch);
  }

  // h_final appended after the [N,OUT] outputs
  for (int i = gtid; i < HDIM; i += NTHR) out[(size_t)NSTEP * OUTDIM + i] = h[i];
}

extern "C" void kernel_launch(void* const* d_in, const int* in_sizes, int n_in,
                              void* d_out, int out_size, void* d_ws,
                              size_t ws_size, hipStream_t stream) {
  (void)in_sizes; (void)n_in; (void)out_size; (void)ws_size;
  // reset grid-barrier counter (graph-capture safe)
  hipMemsetAsync(d_ws, 0, 256, stream);
  ode_rnn_kernel<<<dim3(NWG), dim3(TPB), 0, stream>>>(
      (const float*)d_in[0],  (const float*)d_in[1],  (const float*)d_in[2],
      (const float*)d_in[3],  (const float*)d_in[4],  (const float*)d_in[5],
      (const float*)d_in[6],  (const float*)d_in[7],  (const float*)d_in[8],
      (const float*)d_in[9],  (const float*)d_in[10], (const float*)d_in[11],
      (const float*)d_in[12], (const float*)d_in[13],
      (float*)d_out, (unsigned char*)d_ws);
}
